// MambaCaMix_79061757985306
// MI455X (gfx1250) — matrix operand order repaired and checked by reading, not compiled, not run on hardware
//
#include <hip/hip_runtime.h>
#include <hip/hip_bf16.h>
#include <math.h>

// ---------------------------------------------------------------------------
// Problem constants (from the reference)
// ---------------------------------------------------------------------------
static constexpr int B_   = 2;
static constexpr int CIN_ = 64;
static constexpr int C_   = 128;
static constexpr int H_   = 64;
static constexpr int W_   = 64;
static constexpr int L_   = H_ * W_;      // 4096
static constexpr int DI_  = 256;          // D_INNER
static constexpr int NS_  = 16;           // N_STATE
static constexpr int DTR_ = 8;            // DT_RANK
static constexpr int K_   = 4;            // directions

typedef __attribute__((ext_vector_type(16))) _Float16 v16h;
typedef __attribute__((ext_vector_type(8)))  _Float16 v8h;
typedef __attribute__((ext_vector_type(4)))  _Float16 v4h;
typedef __attribute__((ext_vector_type(8)))  float    v8f;

// ---------------------------------------------------------------------------
// Device helpers
// ---------------------------------------------------------------------------
__device__ __forceinline__ float silu_f(float x) {
  return x / (1.0f + __expf(-x));
}
__device__ __forceinline__ float sigmoid_f(float x) {
  return 1.0f / (1.0f + __expf(-x));
}
__device__ __forceinline__ float softplus_f(float x) {
  return (x > 20.0f) ? x : log1pf(__expf(x));
}

__device__ __forceinline__ float block_sum(float v, float* sbuf) {
  int t = threadIdx.x;
  sbuf[t] = v;
  __syncthreads();
  for (int s = blockDim.x >> 1; s > 0; s >>= 1) {
    if (t < s) sbuf[t] += sbuf[t + s];
    __syncthreads();
  }
  float r = sbuf[0];
  __syncthreads();
  return r;
}

// Two 16B-aligned 8-half runs -> one 16-half fragment (2x global_load_b128).
__device__ __forceinline__ v16h ld16(const _Float16* __restrict__ p0,
                                     const _Float16* __restrict__ p1) {
  v8h a = *(const v8h*)p0;
  v8h b = *(const v8h*)p1;
  return __builtin_shufflevector(a, b, 0, 1, 2, 3, 4, 5, 6, 7,
                                 8, 9, 10, 11, 12, 13, 14, 15);
}

// ---------------------------------------------------------------------------
// Full-tile batched WMMA GEMM on f16 operands, f32 accumulate:
//   C[m,n] = sum_k A[m,k] * W[n,k] (+bias)(+resid)
// Wave computes a 16 x (16*NT) strip; K % 32 == 0, N % (16*NT) == 0
// (weights pre-padded so this always holds).  Inner loop: pure b128 loads +
// v_wmma_f32_16x16x32_f16, zero conversions.
// CDNA5 fragment layouts:
//   A (16x32): row = lane&15; elems 0..7 -> K = 8*kh+e, 8..15 -> K = 16+8*kh+e
//   B (32x16): col = lane&15; same K mapping as A            (kh = lane>>4)
// epi: 0 = none/bias, 2 = softplus(x + bias)
// ---------------------------------------------------------------------------
static constexpr int GEMM_WAVES = 4;

template <int NT, typename OutT>
__global__ void k_gemm_wmma(const _Float16* __restrict__ A,
                            const _Float16* __restrict__ Wm,
                            const float* __restrict__ bias,
                            const float* __restrict__ resid,
                            OutT* __restrict__ Cmat,
                            int M, int N, int Kd,
                            int lda, int ldw, int ldc,
                            long aStride, long cStride,
                            long wStride, int wPeriod, long bStride,
                            int epi) {
  const int lane = threadIdx.x & 31;
  const int z    = blockIdx.z;

  const _Float16* Az = A + (long)z * aStride;
  const _Float16* Wz = Wm + (long)(z % wPeriod) * wStride;
  const float*    Bz = bias ? (bias + (long)(z % wPeriod) * bStride) : nullptr;
  OutT*           Cz = Cmat + (long)z * cStride;
  const float*    Rz = resid ? (resid + (long)z * cStride) : nullptr;

  const int stripsN   = N / (16 * NT);
  const int totStrips = (M >> 4) * stripsN;
  const int strip     = blockIdx.x * GEMM_WAVES + threadIdx.y;
  if (strip >= totStrips) return;               // wave-uniform branch
  const int m0 = (strip / stripsN) << 4;
  const int n0 = (strip % stripsN) * (16 * NT);

  const int kh  = (lane >> 4) << 3;             // A half-select: 0 or 8
  const int kbo = (lane >> 4) << 4;             // unused (0 or 16); B also selects with kh
  const _Float16* arow = Az + (long)(m0 + (lane & 15)) * lda;

  const _Float16* wrow[NT];
#pragma unroll
  for (int t = 0; t < NT; ++t)
    wrow[t] = Wz + (long)(n0 + t * 16 + (lane & 15)) * ldw;

  const v8f vzero = {0.f, 0.f, 0.f, 0.f, 0.f, 0.f, 0.f, 0.f};
  v8f acc[NT];
#pragma unroll
  for (int t = 0; t < NT; ++t) acc[t] = vzero;

  for (int kk = 0; kk < Kd; kk += 32) {
    __builtin_prefetch(arow + kk + 32, 0, 0);   // gfx1250 global_prefetch_b8
    v16h af = ld16(arow + kk + kh, arow + kk + 16 + kh);
#pragma unroll
    for (int t = 0; t < NT; ++t) {
      v16h bf = ld16(wrow[t] + kk + kh, wrow[t] + kk + 16 + kh);
      acc[t] = __builtin_amdgcn_wmma_f32_16x16x32_f16(
          /*neg_a=*/false, af, /*neg_b=*/false, bf,
          /*c_mod=*/(short)0, acc[t], /*reuse_a=*/false, /*reuse_b=*/false);
    }
  }

  const int rbase = m0 + ((lane >> 4) << 3);
#pragma unroll
  for (int t = 0; t < NT; ++t) {
    const int col = n0 + t * 16 + (lane & 15);
    const float bv = Bz ? Bz[col] : 0.0f;
#pragma unroll
    for (int r = 0; r < 8; ++r) {
      float v = acc[t][r] + bv;
      if (epi == 2) v = softplus_f(v);
      long idx = (long)(rbase + r) * ldc + col;
      if (Rz) v += Rz[idx];
      Cz[idx] = (OutT)v;
    }
  }
}

// ---------------------------------------------------------------------------
// Elementwise f32 -> f16 (weights), 4 elems/thread, vectorized
// ---------------------------------------------------------------------------
__global__ void k_cvt_f16(const float* __restrict__ x, _Float16* __restrict__ y,
                          long n) {
  long i = ((long)blockIdx.x * blockDim.x + threadIdx.x) * 4;
  if (i >= n) return;
  float4 v = *(const float4*)(x + i);
  v4h o;
  o[0] = (_Float16)v.x; o[1] = (_Float16)v.y;
  o[2] = (_Float16)v.z; o[3] = (_Float16)v.w;
  *(v4h*)(y + i) = o;
}

// ---------------------------------------------------------------------------
// Weight padding (emit f16 directly)
// ---------------------------------------------------------------------------
// x_proj_w (K,40,256) -> (K,48,256), rows 40..47 = 0
__global__ void k_pad_xproj_w(const float* __restrict__ w, _Float16* __restrict__ wp) {
  long idx = (long)blockIdx.x * blockDim.x + threadIdx.x;
  if (idx >= (long)K_ * 48 * DI_) return;
  int d = (int)(idx & 255);
  int c = (int)((idx >> 8) % 48);
  int k = (int)(idx / (48 * DI_));
  wp[idx] = (c < 40) ? (_Float16)w[((long)(k * 40 + c)) * DI_ + d] : (_Float16)0.0f;
}

// dt_w (K,256,8) -> (K,256,32), cols 8..31 = 0  (K-padded so Kd=32 is exact)
__global__ void k_pad_dt_w(const float* __restrict__ w, _Float16* __restrict__ wp) {
  long idx = (long)blockIdx.x * blockDim.x + threadIdx.x;
  if (idx >= (long)K_ * DI_ * 32) return;
  int r   = (int)(idx & 31);
  long dc = idx >> 5;                  // k*256 + d
  wp[idx] = (r < DTR_) ? (_Float16)w[dc * DTR_ + r] : (_Float16)0.0f;
}

// ---------------------------------------------------------------------------
// K0: transpose x (B,Cin,H,W) -> xt (B, L, Cin) in f16 (WMMA A operand)
// ---------------------------------------------------------------------------
__global__ void k_transpose_in(const float* __restrict__ x, _Float16* __restrict__ xt) {
  long idx = (long)blockIdx.x * blockDim.x + threadIdx.x;
  if (idx >= (long)B_ * L_ * CIN_) return;
  int c = (int)(idx & 63);
  int p = (int)((idx >> 6) & (L_ - 1));
  int b = (int)(idx >> 18);
  xt[idx] = (_Float16)x[(((long)b * CIN_ + c) << 12) + p];
}

// ---------------------------------------------------------------------------
// K2: layernorm over C (f32 in, f16 out = next GEMM's A operand)
// ---------------------------------------------------------------------------
__global__ void k_layernorm(const float* __restrict__ x, const float* __restrict__ g,
                            const float* __restrict__ b, _Float16* __restrict__ y) {
  __shared__ float sbuf[256];
  int row = blockIdx.x;
  int t = threadIdx.x;
  int D = blockDim.x;
  float v  = x[(long)row * D + t];
  float mu = block_sum(v, sbuf) / (float)D;
  float d  = v - mu;
  float var = block_sum(d * d, sbuf) / (float)D;
  y[(long)row * D + t] = (_Float16)(d * rsqrtf(var + 1e-5f) * g[t] + b[t]);
}

// ---------------------------------------------------------------------------
// K4: depthwise 3x3 conv (pad 1) + bias + SiLU; f16 in/out, f32 accumulate
// ---------------------------------------------------------------------------
__global__ void k_dwconv(const _Float16* __restrict__ xz, const float* __restrict__ cw,
                         const float* __restrict__ cb, _Float16* __restrict__ xc) {
  long idx = (long)blockIdx.x * blockDim.x + threadIdx.x;
  if (idx >= (long)B_ * L_ * DI_) return;
  int d = (int)(idx & 255);
  int p = (int)((idx >> 8) & (L_ - 1));
  int b = (int)(idx >> 20);
  int h = p >> 6, w = p & 63;
  float s = cb[d];
#pragma unroll
  for (int dy = -1; dy <= 1; ++dy)
#pragma unroll
    for (int dx = -1; dx <= 1; ++dx) {
      int hh = h + dy, ww = w + dx;
      if (hh >= 0 && hh < H_ && ww >= 0 && ww < W_)
        s += (float)xz[(((long)b * L_ + ((hh << 6) + ww)) * (2 * DI_)) + d] *
             cw[d * 9 + (dy + 1) * 3 + (dx + 1)];
    }
  xc[((long)b * L_ + p) * DI_ + d] = (_Float16)silu_f(s);
}

// ---------------------------------------------------------------------------
// K5: build 4-direction sequences xs (B,K,L,DI) in f16
// ---------------------------------------------------------------------------
__global__ void k_build_xs(const _Float16* __restrict__ xc, _Float16* __restrict__ xs) {
  long idx = (long)blockIdx.x * blockDim.x + threadIdx.x;
  if (idx >= (long)B_ * K_ * L_ * DI_) return;
  int d = (int)(idx & 255);
  int l = (int)((idx >> 8) & (L_ - 1));
  int k = (int)((idx >> 20) & 3);
  int b = (int)(idx >> 22);
  int p;
  if (k == 0)      p = l;
  else if (k == 1) p = ((l & 63) << 6) | (l >> 6);
  else if (k == 2) p = (L_ - 1) - l;
  else { int lf = (L_ - 1) - l; p = ((lf & 63) << 6) | (lf >> 6); }
  xs[idx] = xc[((long)b * L_ + p) * DI_ + d];
}

// ---------------------------------------------------------------------------
// K8: selective scan.  One thread per channel d, 16 f32 states in registers.
// f16 streamed inputs (u, dt, B, C), f32 state math, f32 output.
// xdbl row layout: [0..7]=dts, [8..23]=Bs, [24..39]=Cs (+pad to 48)
// ---------------------------------------------------------------------------
__global__ void k_scan(const _Float16* __restrict__ xs, const _Float16* __restrict__ delta,
                       const _Float16* __restrict__ xdbl,
                       const float* __restrict__ A_logs, const float* __restrict__ Ds,
                       float* __restrict__ yk) {
  int z  = blockIdx.x;                  // (b*K + k)*4 + chunk
  int bk = z >> 2;
  int k  = bk & 3;
  int d  = ((z & 3) << 6) + threadIdx.x;

  const _Float16* u_b  = xs    + (long)bk * L_ * DI_;
  const _Float16* dt_b = delta + (long)bk * L_ * DI_;
  const _Float16* xd_b = xdbl  + (long)bk * L_ * 48;
  float*          y_b  = yk    + (long)bk * L_ * DI_;

  float An[NS_], hst[NS_];
#pragma unroll
  for (int n = 0; n < NS_; ++n) {
    An[n]  = -__expf(A_logs[((long)k * DI_ + d) * NS_ + n]);
    hst[n] = 0.0f;
  }
  const float Dd = Ds[k * DI_ + d];

  for (int l = 0; l < L_; ++l) {
    float dt = (float)dt_b[(long)l * DI_ + d];
    float u  = (float)u_b[(long)l * DI_ + d];
    const _Float16* row = xd_b + (long)l * 48;
    float y = 0.0f;
#pragma unroll
    for (int n = 0; n < NS_; ++n) {
      float dA = __expf(dt * An[n]);
      hst[n] = hst[n] * dA + dt * (float)row[8 + n] * u;
      y += hst[n] * (float)row[24 + n];
    }
    y_b[(long)l * DI_ + d] = y + u * Dd;
  }
}

// ---------------------------------------------------------------------------
// K9: combine 4 directions (index remap), layernorm(DI), gate by silu(z)
// f32 reduction, f16 output (= out_proj's A operand)
// ---------------------------------------------------------------------------
__global__ void k_combine_ln_gate(const float* __restrict__ yk,
                                  const _Float16* __restrict__ xz,
                                  const float* __restrict__ g,
                                  const float* __restrict__ b,
                                  _Float16* __restrict__ ya) {
  __shared__ float sbuf[256];
  int row = blockIdx.x;                 // b*L + p
  int bi  = row >> 12;
  int p   = row & (L_ - 1);
  int d   = threadIdx.x;
  int h = p >> 6, w = p & 63;
  int l1 = (w << 6) | h;                // transposed-direction index
  long base = (long)bi * K_ * L_ * DI_;
  float v = yk[base + ((long)0 * L_ + p) * DI_ + d]
          + yk[base + ((long)1 * L_ + l1) * DI_ + d]
          + yk[base + ((long)2 * L_ + (L_ - 1 - p)) * DI_ + d]
          + yk[base + ((long)3 * L_ + (L_ - 1 - l1)) * DI_ + d];
  float mu  = block_sum(v, sbuf) * (1.0f / DI_);
  float dv  = v - mu;
  float var = block_sum(dv * dv, sbuf) * (1.0f / DI_);
  float yln = dv * rsqrtf(var + 1e-5f) * g[d] + b[d];
  float zv  = (float)xz[(long)row * (2 * DI_) + DI_ + d];
  ya[(long)row * DI_ + d] = (_Float16)(yln * silu_f(zv));
}

// ---------------------------------------------------------------------------
// K11: per-channel mean + max over spatial dims.  mamba layout (B, L, C)
// ---------------------------------------------------------------------------
__global__ void k_chan_stats(const float* __restrict__ mamba,
                             float* __restrict__ avg, float* __restrict__ mx) {
  __shared__ float s1[256];
  __shared__ float s2[256];
  int bc = blockIdx.x;                  // b*C + c
  int b = bc >> 7, c = bc & 127;
  int t = threadIdx.x;
  float sum = 0.0f, mxv = -3.4e38f;
  for (int p = t; p < L_; p += 256) {
    float v = mamba[((long)b * L_ + p) * C_ + c];
    sum += v;
    mxv = fmaxf(mxv, v);
  }
  s1[t] = sum; s2[t] = mxv;
  __syncthreads();
  for (int s = 128; s > 0; s >>= 1) {
    if (t < s) { s1[t] += s1[t + s]; s2[t] = fmaxf(s2[t], s2[t + s]); }
    __syncthreads();
  }
  if (t == 0) { avg[bc] = s1[0] * (1.0f / L_); mx[bc] = s2[0]; }
}

// ---------------------------------------------------------------------------
// K12: channel-attention MLP: att = sigmoid(mlp(avg) + mlp(max)); hidden = 8
// ---------------------------------------------------------------------------
__global__ void k_attention(const float* __restrict__ avg, const float* __restrict__ mx,
                            const float* __restrict__ fc1, const float* __restrict__ fc2,
                            float* __restrict__ att) {
  __shared__ float av[C_];
  __shared__ float mv[C_];
  __shared__ float hid[16];
  int b = blockIdx.x, t = threadIdx.x;
  av[t] = avg[b * C_ + t];
  mv[t] = mx[b * C_ + t];
  __syncthreads();
  if (t < 16) {
    const float* src = (t < 8) ? av : mv;
    int j = t & 7;
    float s = 0.0f;
    for (int c = 0; c < C_; ++c) s += src[c] * fc1[j * C_ + c];
    hid[t] = fmaxf(s, 0.0f);            // relu
  }
  __syncthreads();
  float s = 0.0f;
#pragma unroll
  for (int j = 0; j < 8; ++j) s += (hid[j] + hid[8 + j]) * fc2[t * 8 + j];
  att[b * C_ + t] = sigmoid_f(s);
}

// ---------------------------------------------------------------------------
// K13: group-norm stats over 2 groups of 64 channels x all spatial
// ---------------------------------------------------------------------------
__global__ void k_gn_stats(const float* __restrict__ mamba, const float* __restrict__ att,
                           float* __restrict__ gstat) {
  __shared__ float s1[256];
  __shared__ float s2[256];
  int bg = blockIdx.x;                  // b*2 + g
  int b = bg >> 1, g = bg & 1;
  int t = threadIdx.x;
  float sum = 0.0f, sq = 0.0f;
  const int GN = 64 * L_;
  for (int i = t; i < GN; i += 256) {
    int c = (g << 6) + (i >> 12);
    int p = i & (L_ - 1);
    float v = mamba[((long)b * L_ + p) * C_ + c] * att[b * C_ + c];
    sum += v; sq += v * v;
  }
  s1[t] = sum; s2[t] = sq;
  __syncthreads();
  for (int s = 128; s > 0; s >>= 1) {
    if (t < s) { s1[t] += s1[t + s]; s2[t] += s2[t + s]; }
    __syncthreads();
  }
  if (t == 0) {
    float mu = s1[0] / (float)GN;
    gstat[bg * 2]     = mu;
    gstat[bg * 2 + 1] = s2[0] / (float)GN - mu * mu;
  }
}

// ---------------------------------------------------------------------------
// K14: ca = mamba*att; groupnorm affine; silu; + x1 residual -> out (B,C,H,W)
// ---------------------------------------------------------------------------
__global__ void k_final(const float* __restrict__ mamba, const float* __restrict__ att,
                        const float* __restrict__ gstat,
                        const float* __restrict__ gng, const float* __restrict__ gnb,
                        const float* __restrict__ m, float* __restrict__ out) {
  long idx = (long)blockIdx.x * blockDim.x + threadIdx.x;
  if (idx >= (long)B_ * C_ * L_) return;
  int p = (int)(idx & (L_ - 1));
  int c = (int)((idx >> 12) & 127);
  int b = (int)(idx >> 19);
  float v  = mamba[((long)b * L_ + p) * C_ + c] * att[b * C_ + c];
  int bg   = b * 2 + (c >> 6);
  float mu = gstat[bg * 2], var = gstat[bg * 2 + 1];
  float nr = (v - mu) * rsqrtf(var + 1e-5f) * gng[c] + gnb[c];
  out[idx] = silu_f(nr) + m[((long)b * L_ + p) * C_ + c];
}

// ---------------------------------------------------------------------------
// Host-side orchestration
// ---------------------------------------------------------------------------
extern "C" void kernel_launch(void* const* d_in, const int* in_sizes, int n_in,
                              void* d_out, int out_size, void* d_ws, size_t ws_size,
                              hipStream_t stream) {
  (void)in_sizes; (void)n_in; (void)out_size; (void)ws_size;

  const float* x          = (const float*)d_in[0];
  const float* convin_w   = (const float*)d_in[1];
  const float* convin_b   = (const float*)d_in[2];
  const float* ln_g       = (const float*)d_in[3];
  const float* ln_b       = (const float*)d_in[4];
  const float* in_proj_w  = (const float*)d_in[5];
  const float* conv_w     = (const float*)d_in[6];
  const float* conv_b     = (const float*)d_in[7];
  const float* x_proj_w   = (const float*)d_in[8];
  const float* dt_w       = (const float*)d_in[9];
  const float* dt_b       = (const float*)d_in[10];
  const float* A_logs     = (const float*)d_in[11];
  const float* Ds         = (const float*)d_in[12];
  const float* outnorm_g  = (const float*)d_in[13];
  const float* outnorm_b  = (const float*)d_in[14];
  const float* out_proj_w = (const float*)d_in[15];
  const float* ca_fc1_w   = (const float*)d_in[16];
  const float* ca_fc2_w   = (const float*)d_in[17];
  const float* gn_g       = (const float*)d_in[18];
  const float* gn_b       = (const float*)d_in[19];

  // Bump allocator over d_ws, 256B-aligned blocks
  char*  base = (char*)d_ws;
  size_t off  = 0;
  auto alloc = [&](size_t bytes) -> char* {
    char* p = base + off;
    off += (bytes + 255) & ~(size_t)255;
    return p;
  };

  _Float16* xt_h   = (_Float16*)alloc((size_t)B_ * L_ * CIN_ * 2);       // 1 MB
  float*    m_     = (float*)   alloc((size_t)B_ * L_ * C_ * 4);         // 4 MB
  _Float16* mln_h  = (_Float16*)alloc((size_t)B_ * L_ * C_ * 2);         // 2 MB
  _Float16* xz_h   = (_Float16*)alloc((size_t)B_ * L_ * 2 * DI_ * 2);    // 8 MB
  _Float16* xc_h   = (_Float16*)alloc((size_t)B_ * L_ * DI_ * 2);        // 4 MB
  _Float16* xs_h   = (_Float16*)alloc((size_t)B_ * K_ * L_ * DI_ * 2);   // 16 MB
  _Float16* xdbl_h = (_Float16*)alloc((size_t)B_ * K_ * L_ * 48 * 2);    // 3 MB
  _Float16* dlt_h  = (_Float16*)alloc((size_t)B_ * K_ * L_ * DI_ * 2);   // 16 MB
  float*    yk     = (float*)   alloc((size_t)B_ * K_ * L_ * DI_ * 4);   // 32 MB
  _Float16* ya_h   = (_Float16*)alloc((size_t)B_ * L_ * DI_ * 2);        // 4 MB
  float*    mamba  = (float*)   alloc((size_t)B_ * L_ * C_ * 4);         // 4 MB
  _Float16* wCin   = (_Float16*)alloc((size_t)C_ * CIN_ * 2);
  _Float16* wInP   = (_Float16*)alloc((size_t)2 * DI_ * C_ * 2);
  _Float16* wOutP  = (_Float16*)alloc((size_t)C_ * DI_ * 2);
  _Float16* wpXP   = (_Float16*)alloc((size_t)K_ * 48 * DI_ * 2);
  _Float16* wpDT   = (_Float16*)alloc((size_t)K_ * DI_ * 32 * 2);
  float*    avgb   = (float*)alloc(256 * 4);
  float*    mxb    = (float*)alloc(256 * 4);
  float*    attb   = (float*)alloc(256 * 4);
  float*    gstat  = (float*)alloc(64 * 4);

  const dim3 gblk(32, GEMM_WAVES, 1);

  // Weight conversion / padding (one-shot, all f16 outputs)
  k_cvt_f16<<<(C_ * CIN_ / 4 + 255) / 256, 256, 0, stream>>>(convin_w, wCin, C_ * CIN_);
  k_cvt_f16<<<(2 * DI_ * C_ / 4 + 255) / 256, 256, 0, stream>>>(in_proj_w, wInP, 2 * DI_ * C_);
  k_cvt_f16<<<(C_ * DI_ / 4 + 255) / 256, 256, 0, stream>>>(out_proj_w, wOutP, C_ * DI_);
  k_pad_xproj_w<<<(K_ * 48 * DI_) / 256, 256, 0, stream>>>(x_proj_w, wpXP);
  k_pad_dt_w<<<(K_ * DI_ * 32) / 256, 256, 0, stream>>>(dt_w, wpDT);

  // K0: transpose input to (B, L, Cin) f16
  k_transpose_in<<<2048, 256, 0, stream>>>(x, xt_h);

  // K1: convin 1x1: (8192 x 64) @ (128 x 64)^T + bias -> m (f32)
  k_gemm_wmma<4, float><<<dim3(1024 / GEMM_WAVES, 1, 1), gblk, 0, stream>>>(
      xt_h, wCin, convin_b, nullptr, m_,
      8192, C_, CIN_, CIN_, CIN_, C_,
      0L, 0L, 0L, 1, 0L, 0);

  // K2: layernorm over C -> f16
  k_layernorm<<<B_ * L_, C_, 0, stream>>>(m_, ln_g, ln_b, mln_h);

  // K3: in_proj: (8192 x 128) @ (512 x 128)^T -> xz (f16)
  k_gemm_wmma<4, _Float16><<<dim3(4096 / GEMM_WAVES, 1, 1), gblk, 0, stream>>>(
      mln_h, wInP, nullptr, nullptr, xz_h,
      8192, 2 * DI_, C_, C_, C_, 2 * DI_,
      0L, 0L, 0L, 1, 0L, 0);

  // K4: depthwise 3x3 conv + bias + silu
  k_dwconv<<<(B_ * L_ * DI_) / 256, 256, 0, stream>>>(xz_h, conv_w, conv_b, xc_h);

  // K5: build the 4 scan directions
  k_build_xs<<<(B_ * K_ * L_ * DI_) / 256, 256, 0, stream>>>(xc_h, xs_h);

  // K6: x_proj per (b,k): (4096 x 256) @ (48 x 256)^T -> xdbl (f16)
  k_gemm_wmma<3, _Float16><<<dim3(256 / GEMM_WAVES, 1, B_ * K_), gblk, 0, stream>>>(
      xs_h, wpXP, nullptr, nullptr, xdbl_h,
      L_, 48, DI_, DI_, DI_, 48,
      (long)L_ * DI_, (long)L_ * 48,
      (long)48 * DI_, K_, 0L, 0);

  // K7: dt projection + softplus: K padded to 32; pad cols hit zero weights
  k_gemm_wmma<4, _Float16><<<dim3(1024 / GEMM_WAVES, 1, B_ * K_), gblk, 0, stream>>>(
      xdbl_h, wpDT, dt_b, nullptr, dlt_h,
      L_, DI_, 32, 48, 32, DI_,
      (long)L_ * 48, (long)L_ * DI_,
      (long)DI_ * 32, K_, (long)DI_, 2);

  // K8: selective scan (32 independent blocks, 16 f32 states/thread)
  k_scan<<<B_ * K_ * (DI_ / 64), 64, 0, stream>>>(xs_h, dlt_h, xdbl_h, A_logs, Ds, yk);

  // K9: combine directions + outnorm + gate with silu(z) -> f16
  k_combine_ln_gate<<<B_ * L_, DI_, 0, stream>>>(yk, xz_h, outnorm_g, outnorm_b, ya_h);

  // K10: out_proj + residual m: (8192 x 256) @ (128 x 256)^T + m -> mamba (f32)
  k_gemm_wmma<4, float><<<dim3(1024 / GEMM_WAVES, 1, 1), gblk, 0, stream>>>(
      ya_h, wOutP, nullptr, m_, mamba,
      8192, C_, DI_, DI_, DI_, C_,
      0L, 0L, 0L, 1, 0L, 0);

  // K11-K12: channel attention
  k_chan_stats<<<B_ * C_, 256, 0, stream>>>(mamba, avgb, mxb);
  k_attention<<<B_, C_, 0, stream>>>(avgb, mxb, ca_fc1_w, ca_fc2_w, attb);

  // K13-K14: group norm + silu + residual x1 -> output (B, C, H, W)
  k_gn_stats<<<B_ * 2, 256, 0, stream>>>(mamba, attb, gstat);
  k_final<<<(B_ * C_ * L_) / 256, 256, 0, stream>>>(mamba, attb, gstat, gn_g, gn_b,
                                                    m_, (float*)d_out);
}
